// AngularLocalSensitiveHashing_31585189495342
// MI455X (gfx1250) — compile-verified, matrix operand order checked by reading, not run on hardware
//
#include <hip/hip_runtime.h>
#include <hip/hip_bf16.h>

#define R_ROUNDS 8
#define BATCHSZ  8
#define LEN      4096
#define HID      128
#define NH       64     // N_HASHES = MAX_LEN / BUCKET
#define STRIPS   8      // 16-row strips processed per wave (B reuse factor)

typedef __bf16 v16bf __attribute__((ext_vector_type(16)));
typedef float  v8f   __attribute__((ext_vector_type(8)));
typedef float  v4f   __attribute__((ext_vector_type(4)));

struct WaveLds {
    float xt[16][128];   // x tile (f32); columns 0..63 reused as rot[] later
    float invn[16];      // per-row 1/norm
    float partial[32];   // norm partial sums
};

// ---------------------------------------------------------------------------
// Kernel 1: normalized rotation via bf16 WMMA + per-row argmax -> hash bucket.
// One block = 4 waves, one (r,b); B (rm[r]) converted to bf16 fragments in LDS
// ONCE per block, then each wave sweeps 8 strips of 16 rows reusing it.
// ---------------------------------------------------------------------------
__global__ __launch_bounds__(128) void lsh_hash_wmma(
    const float* __restrict__ x,        // (B, L, D) f32
    const float* __restrict__ rm,       // (16, NH, HID) f32
    unsigned int* __restrict__ hashes)  // (R, B, L) u32 in [1,128]
{
    __shared__ WaveLds lds[4];
    __shared__ v16bf ldsB[4][4][32];    // [ct][kb][lane] bf16 B fragments, 16 KB

    const int tid  = threadIdx.x;
    const int wave = tid >> 5;
    const int lane = tid & 31;
    WaveLds& w = lds[wave];

    const unsigned groupsPerRB = (LEN / 16u) / STRIPS;          // 32
    const unsigned group  = blockIdx.x * 4u + (unsigned)wave;   // 0..R*B*32-1
    const unsigned l0base = (group % groupsPerRB) * (16u * STRIPS);
    const unsigned rb     = group / groupsPerRB;                // uniform/block
    const unsigned b      = rb % BATCHSZ;
    const unsigned r      = rb / BATCHSZ;

    // ---- Stage B fragments once per block: ldsB[ct][kb][ln][i] =
    //      bf16( rm[r][ct*16 + (ln&15)][kb*32 + 16*(ln>=16) + i] )
    {
        const float* rmr = rm + (size_t)r * NH * HID;
        const int e0 = tid * 64;                    // 8192 entries / 128 thr
        #pragma unroll
        for (int u = 0; u < 64; ++u) {
            const int e  = e0 + u;
            const int i  = e & 15;
            const int ln = (e >> 4) & 31;
            const int kb = (e >> 9) & 3;
            const int ct = (e >> 11) & 3;
            const int c  = ct * 16 + (ln & 15);
            const int k  = kb * 32 + ((ln < 16) ? 0 : 16) + i;
            ldsB[ct][kb][ln][i] = (__bf16)rmr[(size_t)c * HID + k];
        }
    }
    __syncthreads();

    const int am    = lane & 15;               // A row within strip
    const int khalf = (lane < 16) ? 0 : 16;    // K sub-block per lane half

    for (int s = 0; s < STRIPS; ++s) {
        const unsigned l0 = l0base + (unsigned)s * 16u;

        // ---- Phase A: stage 16x128 f32 tile of x into LDS (128-bit loads)
        {
            const int row  = lane >> 1;        // 0..15
            const int half = lane & 1;         // 64 contiguous floats each
            const v4f* src = reinterpret_cast<const v4f*>(
                x + ((size_t)b * LEN + (l0 + row)) * HID + half * 64);
            v4f* dst = reinterpret_cast<v4f*>(&w.xt[row][half * 64]);
            #pragma unroll
            for (int i = 0; i < 16; ++i) dst[i] = src[i];
        }
        __syncthreads();

        // ---- Phase B: row norms (2 lanes per row, combine halves)
        {
            const int row  = lane & 15;
            const int half = lane >> 4;
            float ssum = 0.f;
            #pragma unroll
            for (int i = 0; i < 64; ++i) {
                float v = w.xt[row][half * 64 + i];
                ssum += v * v;
            }
            w.partial[lane] = ssum;
        }
        __syncthreads();
        if (lane < 16) {
            float ssum = w.partial[lane] + w.partial[lane + 16];
            float nrm  = fmaxf(sqrtf(ssum), 1e-12f);
            w.invn[lane] = 1.0f / nrm;
        }
        __syncthreads();

        // ---- Phase C: normalized bf16 A fragments (4 k-steps of 16x16x32)
        v16bf afrag[4];
        {
            const float scale = w.invn[am];
            #pragma unroll
            for (int kb = 0; kb < 4; ++kb) {
                #pragma unroll
                for (int i = 0; i < 16; ++i) {
                    afrag[kb][i] =
                        (__bf16)(w.xt[am][kb * 32 + khalf + i] * scale);
                }
            }
        }
        __syncthreads();

        // ---- Phase D: rot tile = A(16x128) x RM[r]^T; rot overlays xt[:, :64]
        #pragma unroll
        for (int ct = 0; ct < 4; ++ct) {
            v8f acc = {};
            #pragma unroll
            for (int kb = 0; kb < 4; ++kb) {
                v16bf bfrag = ldsB[ct][kb][lane];
                acc = __builtin_amdgcn_wmma_f32_16x16x32_bf16(
                    false, afrag[kb], false, bfrag, (short)0, acc,
                    false, false);
            }
            // C/D layout: VGPR j -> (M = j + 8*(lane>=16), N = lane&15)
            const int rbase = (lane < 16) ? 0 : 8;
            #pragma unroll
            for (int j = 0; j < 8; ++j) {
                w.xt[rbase + j][ct * 16 + am] = acc[j];   // rot value
            }
        }
        __syncthreads();

        // ---- Phase E: argmax over concat([rot, -rot]) per row (first wins)
        if (lane < 16) {
            const int row = lane;
            float best = -3.4e38f;
            int   arg  = 0;
            #pragma unroll 4
            for (int j = 0; j < 128; ++j) {
                float v = (j < 64) ? w.xt[row][j] : -w.xt[row][j - 64];
                if (v > best) { best = v; arg = j; }
            }
            hashes[((size_t)r * BATCHSZ + b) * LEN + l0 + row] =
                (unsigned)(arg + 1);
        }
        __syncthreads();
    }
}

// ---------------------------------------------------------------------------
// Kernel 2: per-(r,b) in-LDS bitonic sort of packed keys hash*4096 + pos.
// Keys are unique, so plain sort == stable sort-by-(hash,pos).
// ---------------------------------------------------------------------------
__global__ __launch_bounds__(256) void lsh_sort(
    const unsigned int* __restrict__ hashes, // (R,B,L)
    int* __restrict__ out_hashes,            // (R,B,L)
    int* __restrict__ out_indices)           // (R,B,L)
{
    __shared__ unsigned int key[LEN];        // 16 KB
    const int tid = threadIdx.x;
    const unsigned rb = blockIdx.x;
    const unsigned int* h = hashes + (size_t)rb * LEN;

    for (int i = tid; i < LEN; i += 256)
        key[i] = h[i] * (unsigned)LEN + (unsigned)i;
    __syncthreads();

    for (unsigned k = 2; k <= (unsigned)LEN; k <<= 1) {
        for (unsigned j = k >> 1; j > 0; j >>= 1) {
            for (unsigned i = tid; i < (unsigned)LEN; i += 256) {
                unsigned ixj = i ^ j;
                if (ixj > i) {
                    unsigned a = key[i], c = key[ixj];
                    bool up = ((i & k) == 0);
                    if ((a > c) == up) { key[i] = c; key[ixj] = a; }
                }
            }
            __syncthreads();
        }
    }

    for (int i = tid; i < LEN; i += 256) {
        unsigned v = key[i];
        out_hashes [(size_t)rb * LEN + i] = (int)(v >> 12);     // hash
        out_indices[(size_t)rb * LEN + i] = (int)(v & 4095u);   // position
    }
}

// ---------------------------------------------------------------------------
// Kernel 3: bandwidth-bound gather. One wave per output row: 32 lanes x
// 128-bit = 512 B. x (16 MB) stays L2-resident; output stream is non-temporal.
// ---------------------------------------------------------------------------
__global__ __launch_bounds__(256) void lsh_gather(
    const float* __restrict__ x,       // (B,L,D)
    const int*   __restrict__ indices, // (R,B,L)
    float*       __restrict__ xs)      // (R,B,L,D) flat
{
    const int tid  = threadIdx.x;
    const int wave = tid >> 5;
    const int lane = tid & 31;
    const size_t   row = (size_t)blockIdx.x * 8 + (size_t)wave; // 0..R*B*L-1
    const unsigned rb  = (unsigned)(row >> 12);
    const unsigned b   = rb % BATCHSZ;
    const int idx = indices[row];

    const v4f* src = reinterpret_cast<const v4f*>(
        x + ((size_t)b * LEN + (unsigned)idx) * HID) + lane;
    v4f* dst = reinterpret_cast<v4f*>(xs + row * HID) + lane;
    v4f val = *src;
    __builtin_nontemporal_store(val, dst);
}

extern "C" void kernel_launch(void* const* d_in, const int* in_sizes, int n_in,
                              void* d_out, int out_size, void* d_ws, size_t ws_size,
                              hipStream_t stream) {
    const float* x  = (const float*)d_in[0];   // (8, 4096, 128) f32
    const float* rm = (const float*)d_in[1];   // (16, 64, 128) f32

    unsigned int* hashes = (unsigned int*)d_ws;           // R*B*L u32 = 1 MiB

    // d_out layout (flat concat, return order):
    //   x_sorted:      R*B*L*D f32
    //   sorted_hashes: R*B*L   i32
    //   sorted_indices:R*B*L   i32
    float* xs   = (float*)d_out;
    int* out_h  = (int*)d_out + (size_t)R_ROUNDS * BATCHSZ * LEN * HID;
    int* out_i  = out_h + (size_t)R_ROUNDS * BATCHSZ * LEN;

    // 1) hashes via WMMA: 512 blocks x 4 waves, 8 strips per wave
    lsh_hash_wmma<<<R_ROUNDS * BATCHSZ * ((LEN / 16) / STRIPS) / 4, 128, 0,
                    stream>>>(x, rm, hashes);

    // 2) per-(r,b) bitonic sort of packed keys
    lsh_sort<<<R_ROUNDS * BATCHSZ, 256, 0, stream>>>(hashes, out_h, out_i);

    // 3) gather rows (streaming, non-temporal output)
    lsh_gather<<<R_ROUNDS * BATCHSZ * LEN / 8, 256, 0, stream>>>(x, out_i, xs);
}
